// SpikeLoss_28097676051365
// MI455X (gfx1250) — matrix operand order repaired and checked by reading, not compiled, not run on hardware
//
#include <hip/hip_runtime.h>

typedef __attribute__((ext_vector_type(16))) _Float16 v16h;
typedef __attribute__((ext_vector_type(8)))  float    v8f;
typedef __attribute__((ext_vector_type(4)))  float    v4f;

#define T_STEPS   32
#define TILE_ROWS 16

// Main kernel: each wave32 processes 16-row x 32-timestep tiles.
// psp is computed with V_WMMA_F32_16X16X32_F16 (A = f16 spikes, exact; C/D = f32).
__global__ void __launch_bounds__(256)
spike_loss_wmma_kernel(const float* __restrict__ outputs,
                       const float* __restrict__ target,
                       const int*   __restrict__ tau_p,
                       float*       __restrict__ partials,
                       int nTiles)
{
    const int lane        = threadIdx.x & 31;
    const int waveInBlock = threadIdx.x >> 5;
    const int wavesPerBlk = blockDim.x >> 5;
    const int waveGlobal  = blockIdx.x * wavesPerBlk + waveInBlock;
    const int nWaves      = gridDim.x * wavesPerBlk;

    const float tau     = (float)tau_p[0];
    const float decay   = 1.0f - 1.0f / tau;
    const float inv_tau = 1.0f / tau;

    const int hi    = lane >> 4;   // 0 for lanes 0-15, 1 for lanes 16-31
    const int lo    = lane & 15;
    const int kbase = hi * 16;     // this lane's K range in the B matrix

    // Build the two constant B tiles (f16): M[k][n] = decay^(n-k)/tau for k<=n.
    // B k order chosen here (contiguous, 32x16): lanes 0-15 hold K=0..15, lanes 16-31 hold K=16..31,
    // column n = lane&15; half j within the 8 VGPRs corresponds to K = kbase + j.
    v16h b0, b1;
    for (int j = 0; j < 16; ++j) {
        const int k  = kbase + j;
        const int e0 = lo - k;        // exponent for time column n = lo
        const int e1 = lo + 16 - k;   // exponent for time column n = lo + 16
        float p0 = 0.0f, p1 = 0.0f;
        if (e0 >= 0) { p0 = inv_tau; for (int q = 0; q < e0; ++q) p0 *= decay; }
        if (e1 >= 0) { p1 = inv_tau; for (int q = 0; q < e1; ++q) p1 *= decay; }
        b0[j] = (_Float16)p0;
        b1[j] = (_Float16)p1;
    }

    float acc = 0.0f;

    for (int tile = waveGlobal; tile < nTiles; tile += nWaves) {
        const long rowBase = (long)tile * TILE_ROWS;

        // ---- Load A (target) tile in the same contiguous k order as B.
        // Lane (hi,lo) owns matrix row M = lo.
        //   hi==0: halves 0-15 = cols 0-15
        //   hi==1: halves 0-15 = cols 16-31
        const float* trow = target + (rowBase + lo) * T_STEPS;
        const int    cb   = hi * 16;
        const v4f f0 = *(const v4f*)(trow + cb);
        const v4f f1 = *(const v4f*)(trow + cb + 4);
        const v4f f2 = *(const v4f*)(trow + cb + 8);
        const v4f f3 = *(const v4f*)(trow + cb + 12);

        v16h a;
        a[0]  = (_Float16)f0[0]; a[1]  = (_Float16)f0[1];
        a[2]  = (_Float16)f0[2]; a[3]  = (_Float16)f0[3];
        a[4]  = (_Float16)f1[0]; a[5]  = (_Float16)f1[1];
        a[6]  = (_Float16)f1[2]; a[7]  = (_Float16)f1[3];
        a[8]  = (_Float16)f2[0]; a[9]  = (_Float16)f2[1];
        a[10] = (_Float16)f2[2]; a[11] = (_Float16)f2[3];
        a[12] = (_Float16)f3[0]; a[13] = (_Float16)f3[1];
        a[14] = (_Float16)f3[2]; a[15] = (_Float16)f3[3];

        // ---- psp tiles: D = A x B (+0), f32 accumulate.
        v8f p0 = {};
        v8f p1 = {};
        p0 = __builtin_amdgcn_wmma_f32_16x16x32_f16(false, a, false, b0,
                                                    (short)0, p0, false, false);
        p1 = __builtin_amdgcn_wmma_f32_16x16x32_f16(false, a, false, b1,
                                                    (short)0, p1, false, false);

        // ---- delta^2 accumulation, reading outputs per the f32 C/D layout:
        // VGPR r <-> row (r + 8*hi), column lo (p0) / lo+16 (p1).
        const float* obase = outputs + rowBase * T_STEPS;
        #pragma unroll
        for (int r = 0; r < 8; ++r) {
            const float* orow = obase + (long)(r + hi * 8) * T_STEPS + lo;
            const float d0 = orow[0]  - p0[r];
            const float d1 = orow[16] - p1[r];
            acc = fmaf(d0, d0, acc);
            acc = fmaf(d1, d1, acc);
        }
    }

    // ---- wave32 reduction, then block reduction via LDS.
    #pragma unroll
    for (int off = 16; off > 0; off >>= 1)
        acc += __shfl_down(acc, off, 32);

    __shared__ float sdata[8];
    if (lane == 0) sdata[waveInBlock] = acc;
    __syncthreads();
    if (threadIdx.x == 0) {
        float s = 0.0f;
        for (int w = 0; w < wavesPerBlk; ++w) s += sdata[w];
        partials[blockIdx.x] = s;
    }
}

// Deterministic final reduction: sum per-block partials, apply the 0.5 factor.
__global__ void __launch_bounds__(256)
spike_loss_reduce_kernel(const float* __restrict__ partials, int n,
                         float* __restrict__ out)
{
    __shared__ float sdata[256];
    float s = 0.0f;
    for (int i = threadIdx.x; i < n; i += blockDim.x) s += partials[i];
    sdata[threadIdx.x] = s;
    __syncthreads();
    for (int stride = 128; stride > 0; stride >>= 1) {
        if ((int)threadIdx.x < stride) sdata[threadIdx.x] += sdata[threadIdx.x + stride];
        __syncthreads();
    }
    if (threadIdx.x == 0) out[0] = 0.5f * sdata[0];
}

extern "C" void kernel_launch(void* const* d_in, const int* in_sizes, int n_in,
                              void* d_out, int out_size, void* d_ws, size_t ws_size,
                              hipStream_t stream) {
    const float* outputs = (const float*)d_in[0];
    const float* target  = (const float*)d_in[1];
    // d_in[2] = n_steps (assumed 32; tile layout is specialized for T=32)
    const int*   tau_p   = (const int*)d_in[3];

    const int rows   = in_sizes[0] / T_STEPS;   // B*C*H*W
    const int nTiles = rows / TILE_ROWS;

    const int blocks  = 1024;
    const int threads = 256;   // 8 wave32 waves per block
    float* partials = (float*)d_ws;             // 1024 floats of scratch

    spike_loss_wmma_kernel<<<blocks, threads, 0, stream>>>(
        outputs, target, tau_p, partials, nTiles);
    spike_loss_reduce_kernel<<<1, threads, 0, stream>>>(
        partials, blocks, (float*)d_out);
}
